// Attention_40492951666725
// MI455X (gfx1250) — matrix operand order repaired and checked by reading, not compiled, not run on hardware
//
#include <hip/hip_runtime.h>
#include <math.h>

// ---------------------------------------------------------------------------
// Problem constants (from reference): B=4, N=2048, DIM=1024, H=16, DH=64,
// ROT=32, INNER=1024.  All GEMM stages run as BF16 WMMA (16x16x32, f32 acc).
// ---------------------------------------------------------------------------
#define B_     4
#define N_     2048
#define DIM_   1024
#define H_     16
#define DH_    64
#define ROT_   32
#define INNER_ 1024
#define BH_    (B_ * H_)

typedef __attribute__((ext_vector_type(16))) __bf16 v16bf;
typedef __attribute__((ext_vector_type(8)))  float  v8f;

__device__ __forceinline__ __bf16 f2bf(float f) { return (__bf16)f; }

__device__ __forceinline__ v8f wmma_bf16(v16bf a, v16bf b, v8f c) {
  // D = A(16x32 bf16) * B(32x16 bf16) + C(16x16 f32)
  return __builtin_amdgcn_wmma_f32_16x16x32_bf16(
      /*neg_a=*/false, a, /*neg_b=*/false, b,
      /*c_mod=*/(short)0, c, /*reuse_a=*/false, /*reuse_b=*/false);
}

// CDNA5 async DMA: cache/memory -> LDS, 16B per lane, tracked with ASYNCcnt.
// VDST VGPR carries the per-lane LDS byte address (ISA 15.18.3 op 98).
__device__ __forceinline__ void async_b128(uint32_t lds_addr, const void* gaddr) {
  asm volatile("global_load_async_to_lds_b128 %0, %1, off"
               :: "v"(lds_addr), "v"(gaddr)
               : "memory");
}
__device__ __forceinline__ void wait_async_le4() {
  asm volatile("s_wait_asynccnt 0x4" ::: "memory");
}
__device__ __forceinline__ uint32_t lds_off(const void* p) {
  return (uint32_t)(uintptr_t)p;
}

// A-fragment loader: 16x32 bf16 tile, row-major in LDS with row stride STRIDE.
// Lanes 0-15 hold row M=lane, K=0..15; lanes 16-31 hold
// row M=lane-16, K=16..31; 2 packed bf16 per VGPR.
template <int STRIDE>
__device__ __forceinline__ v16bf load_a_frag(const __bf16* base, int lane) {
  const int half = lane >> 4, mr = lane & 15;
  const __bf16* p = base + mr * STRIDE;
  v16bf a;
#pragma unroll
  for (int e = 0; e < 16; ++e) {
    const int k = half * 16 + e;
    a[e] = p[k];
  }
  return a;
}

// B-fragment loader: 32x16 bf16 tile (K-major rows) with row stride STRIDE.
// Lanes 0-15 hold K=0..15 of column N=lane; lanes 16-31 hold K=16..31.
template <int STRIDE>
__device__ __forceinline__ v16bf load_b_frag(const __bf16* base, int lane) {
  const int half = lane >> 4, nc = lane & 15;
  v16bf b;
#pragma unroll
  for (int e = 0; e < 16; ++e) b[e] = base[(half * 16 + e) * STRIDE + nc];
  return b;
}

// ---------------------------------------------------------------------------
// Kernel 1: qkv = x(8192x1024) @ Wqkv(1024x3072), BF16 WMMA, epilogue scatters
// into head-major q/k/v [BH][N][DH] bf16 and folds DH^-0.5 into q.
// Block = 128 threads (4 waves), tile 64x64, K-step 32.
// ---------------------------------------------------------------------------
#define LDA1 34   // 32 + 2  (68B rows -> 17-bank stride, conflict free)
#define LDB1 66   // 64 + 2

__global__ __launch_bounds__(128) void qkv_gemm_kernel(
    const float* __restrict__ X, const float* __restrict__ W,
    __bf16* __restrict__ qh, __bf16* __restrict__ kh, __bf16* __restrict__ vh) {
  __shared__ __bf16 As[64 * LDA1];
  __shared__ __bf16 Bs[32 * LDB1];
  const int tid = threadIdx.x, lane = tid & 31, wave = tid >> 5;
  const int m0 = blockIdx.y * 64, n0 = blockIdx.x * 64;

  v8f acc[4] = {v8f{}, v8f{}, v8f{}, v8f{}};

  for (int k0 = 0; k0 < DIM_; k0 += 32) {
    // L2 prefetch of the next K-tile (global_prefetch_b8)
    if (k0 + 32 < DIM_) {
      const int ia = (tid * 4) >> 5, ib = (tid * 4) >> 6;
      __builtin_prefetch(X + (size_t)(m0 + ia) * DIM_ + k0 + 32, 0, 1);
      __builtin_prefetch(W + (size_t)(k0 + 32 + ib) * (3 * INNER_) + n0, 0, 1);
    }
    // stage A: 64x32 f32 -> bf16
#pragma unroll
    for (int i = 0; i < 4; ++i) {
      const int idx = (tid + i * 128) * 4;
      const int r = idx >> 5, c = idx & 31;
      const float4 v = *(const float4*)(X + (size_t)(m0 + r) * DIM_ + k0 + c);
      __bf16* d = &As[r * LDA1 + c];
      d[0] = f2bf(v.x); d[1] = f2bf(v.y); d[2] = f2bf(v.z); d[3] = f2bf(v.w);
    }
    // stage B: 32x64 f32 -> bf16
#pragma unroll
    for (int i = 0; i < 4; ++i) {
      const int idx = (tid + i * 128) * 4;
      const int r = idx >> 6, c = idx & 63;
      const float4 v = *(const float4*)(W + (size_t)(k0 + r) * (3 * INNER_) + n0 + c);
      __bf16* d = &Bs[r * LDB1 + c];
      d[0] = f2bf(v.x); d[1] = f2bf(v.y); d[2] = f2bf(v.z); d[3] = f2bf(v.w);
    }
    __syncthreads();
    const v16bf a = load_a_frag<LDA1>(&As[(wave * 16) * LDA1], lane);
#pragma unroll
    for (int cb = 0; cb < 4; ++cb) {
      const v16bf b = load_b_frag<LDB1>(&Bs[cb * 16], lane);
      acc[cb] = wmma_bf16(a, b, acc[cb]);
    }
    __syncthreads();
  }

  // epilogue: scatter into q/k/v head-major bf16
  const int half = lane >> 4, nc = lane & 15;
#pragma unroll
  for (int cb = 0; cb < 4; ++cb) {
    const int gc = n0 + cb * 16 + nc;
    const int which = gc >> 10;            // 0=q 1=k 2=v
    const int hh = (gc & 1023) >> 6;
    const int dd = gc & 63;
    __bf16* dst = (which == 0) ? qh : (which == 1) ? kh : vh;
    const float sc = (which == 0) ? 0.125f : 1.0f;   // DH^-0.5 = 1/8
#pragma unroll
    for (int i = 0; i < 8; ++i) {
      const int gr = m0 + wave * 16 + i + half * 8;  // = b*N + n
      const int bb = gr >> 11, nn = gr & (N_ - 1);
      const size_t off = ((size_t)(bb * H_ + hh) * N_ + nn) * DH_ + dd;
      dst[off] = f2bf(acc[cb][i] * sc);
    }
  }
}

// ---------------------------------------------------------------------------
// Kernel 2: in-place RoPE on first ROT dims of q and k (head-major bf16).
// One thread per (q|k, bh, n, even/odd pair) -> race-free in place.
// ---------------------------------------------------------------------------
__global__ void rope_kernel(__bf16* __restrict__ qh, __bf16* __restrict__ kh,
                            const float* __restrict__ sinT,
                            const float* __restrict__ cosT) {
  int idx = blockIdx.x * blockDim.x + threadIdx.x;
  const int total = 2 * BH_ * N_ * (ROT_ / 2);
  if (idx >= total) return;
  const int p  = idx & 15;            idx >>= 4;
  const int n  = idx & (N_ - 1);      idx >>= 11;
  const int bh = idx & (BH_ - 1);     idx >>= 6;
  __bf16* t = (idx == 0) ? qh : kh;
  const size_t base = ((size_t)bh * N_ + n) * DH_ + p * 2;
  const float e0 = (float)t[base], e1 = (float)t[base + 1];
  const float c0 = cosT[n * ROT_ + p * 2], c1 = cosT[n * ROT_ + p * 2 + 1];
  const float s0 = sinT[n * ROT_ + p * 2], s1 = sinT[n * ROT_ + p * 2 + 1];
  t[base]     = f2bf(e0 * c0 - e1 * s0);   // rotate_every_two: even gets -odd
  t[base + 1] = f2bf(e1 * c1 + e0 * s1);   // odd gets +even
}

// ---------------------------------------------------------------------------
// Kernel 3: flash attention. Block = (head, 64 query rows), 4 waves, each wave
// owns 16 rows. K/V tiles stream through double-buffered LDS via CDNA5
// GLOBAL_LOAD_ASYNC_TO_LDS_B128 (ASYNCcnt DMA) so tile t+1 DMA overlaps the
// WMMA work on tile t. Per wave each tile = exactly 4 async ops, so
// s_wait_asynccnt 4 (after issuing the prefetch) retires the current tile and,
// in-order, the initial Q staging. Output -> ao[B*N][INNER] bf16.
// ---------------------------------------------------------------------------
#define TJ   32
#define LDQ  (DH_ + 8)   // 72 elems = 144B rows (16B aligned, 36-bank stride)
#define LDP  (TJ + 8)    // 40 elems = 80B rows

__global__ __launch_bounds__(128) void attn_kernel(
    const __bf16* __restrict__ qh, const __bf16* __restrict__ kh,
    const __bf16* __restrict__ vh, __bf16* __restrict__ ao) {
  __shared__ __bf16 qs[64 * LDQ];
  __shared__ __bf16 ks[2][TJ * LDQ];
  __shared__ __bf16 vs[2][TJ * LDQ];
  __shared__ __bf16 ps[4][16 * LDP];

  const int tid = threadIdx.x, lane = tid & 31, wave = tid >> 5;
  const int half = lane >> 4, nc = lane & 15;
  const int bh = blockIdx.y;
  const int q0 = blockIdx.x * 64;
  const __bf16* qp = qh + ((size_t)bh * N_ + q0) * DH_;
  const __bf16* kp = kh + (size_t)bh * N_ * DH_;
  const __bf16* vp = vh + (size_t)bh * N_ * DH_;

  // per-thread copy slots (8 bf16 = 16B each)
  const int idxA = tid * 8, idxB = (tid + 128) * 8;
  const int rA = idxA >> 6, cA = idxA & 63;
  const int rB = idxB >> 6, cB = idxB & 63;

  // ---- stage Q block 64x64 via async DMA (4 ops/thread)
  {
    const int i2 = (tid + 256) * 8, i3 = (tid + 384) * 8;
    const int r2 = i2 >> 6, c2 = i2 & 63, r3 = i3 >> 6, c3 = i3 & 63;
    async_b128(lds_off(&qs[rA * LDQ + cA]), &qp[(size_t)rA * DH_ + cA]);
    async_b128(lds_off(&qs[rB * LDQ + cB]), &qp[(size_t)rB * DH_ + cB]);
    async_b128(lds_off(&qs[r2 * LDQ + c2]), &qp[(size_t)r2 * DH_ + c2]);
    async_b128(lds_off(&qs[r3 * LDQ + c3]), &qp[(size_t)r3 * DH_ + c3]);
  }

  // ---- K/V tile DMA: exactly 4 async ops per thread per tile
  auto issue_kv = [&](int j0, int buf) {
    async_b128(lds_off(&ks[buf][rA * LDQ + cA]), &kp[(size_t)(j0 + rA) * DH_ + cA]);
    async_b128(lds_off(&vs[buf][rA * LDQ + cA]), &vp[(size_t)(j0 + rA) * DH_ + cA]);
    async_b128(lds_off(&ks[buf][rB * LDQ + cB]), &kp[(size_t)(j0 + rB) * DH_ + cB]);
    async_b128(lds_off(&vs[buf][rB * LDQ + cB]), &vp[(size_t)(j0 + rB) * DH_ + cB]);
  };
  issue_kv(0, 0);

  v8f o[4] = {v8f{}, v8f{}, v8f{}, v8f{}};
  float m[8], ssum[8];
#pragma unroll
  for (int i = 0; i < 8; ++i) { m[i] = -1e30f; ssum[i] = 0.f; }

  for (int j0 = 0, it = 0; j0 < N_; j0 += TJ, ++it) {
    const int cur = it & 1, nxt = cur ^ 1;
    __syncthreads();                        // readers of buf `nxt` are done
    const int jn = (j0 + TJ) & (N_ - 1);    // last iter wraps: harmless dummy
    issue_kv(jn, nxt);                      // overlap DMA with compute below
    wait_async_le4();                       // current tile (and Q) landed
    __syncthreads();                        // ...for every wave in the group
    const __bf16* ksc = ks[cur];
    const __bf16* vsc = vs[cur];

    // ---- S = Q(16x64) @ K^T(64x32): two 16x16 accumulators, K in 32-chunks
    v8f s0 = v8f{}, s1 = v8f{};
#pragma unroll
    for (int kk = 0; kk < DH_; kk += 32) {
      const v16bf a = load_a_frag<LDQ>(&qs[(wave * 16) * LDQ + kk], lane);
      // B[k][n] = K^T[kk+k][j] = ks[j][kk+k] : per-lane contiguous 16 bf16
      v16bf b0, b1;
      const __bf16* kb0 = &ksc[nc * LDQ + kk + half * 16];
      const __bf16* kb1 = &ksc[(16 + nc) * LDQ + kk + half * 16];
#pragma unroll
      for (int e = 0; e < 16; ++e) { b0[e] = kb0[e]; b1[e] = kb1[e]; }
      s0 = wmma_bf16(a, b0, s0);
      s1 = wmma_bf16(a, b1, s1);
    }

    // ---- online softmax (rows striped: VGPR i <-> row i + 8*half)
    float pl[8], ph[8], corr[8];
#pragma unroll
    for (int i = 0; i < 8; ++i) {
      float t = fmaxf(s0[i], s1[i]);
      t = fmaxf(t, __shfl_xor(t, 1, 32));
      t = fmaxf(t, __shfl_xor(t, 2, 32));
      t = fmaxf(t, __shfl_xor(t, 4, 32));
      t = fmaxf(t, __shfl_xor(t, 8, 32));
      const float mn = fmaxf(m[i], t);
      const float p0 = __expf(s0[i] - mn), p1 = __expf(s1[i] - mn);
      float rs = p0 + p1;
      rs += __shfl_xor(rs, 1, 32);
      rs += __shfl_xor(rs, 2, 32);
      rs += __shfl_xor(rs, 4, 32);
      rs += __shfl_xor(rs, 8, 32);
      const float c = __expf(m[i] - mn);
      ssum[i] = ssum[i] * c + rs;
      m[i] = mn;
      corr[i] = c; pl[i] = p0; ph[i] = p1;
    }
#pragma unroll
    for (int cb = 0; cb < 4; ++cb)
#pragma unroll
      for (int i = 0; i < 8; ++i) o[cb][i] *= corr[i];

    // ---- bounce P (16x32 bf16) through per-wave LDS to get A-frag layout
    __bf16* pw = ps[wave];
#pragma unroll
    for (int i = 0; i < 8; ++i) {
      const int r = i + 8 * half;
      pw[r * LDP + nc]      = f2bf(pl[i]);
      pw[r * LDP + 16 + nc] = f2bf(ph[i]);
    }
    const v16bf pa = load_a_frag<LDP>(pw, lane);  // wave-local; DScnt ordered

    // ---- O += P(16x32) @ V(32x64)
#pragma unroll
    for (int cb = 0; cb < 4; ++cb) {
      v16bf bv;
      const __bf16* vb = &vsc[(half * 16) * LDQ + cb * 16 + nc];
#pragma unroll
      for (int e = 0; e < 16; ++e) bv[e] = vb[e * LDQ];
      o[cb] = wmma_bf16(pa, bv, o[cb]);
    }
  }

  // ---- epilogue: O /= l, write (B,N,H*DH) bf16
  const int b = bh >> 4, h = bh & 15;
#pragma unroll
  for (int cb = 0; cb < 4; ++cb)
#pragma unroll
    for (int i = 0; i < 8; ++i) {
      const int r = wave * 16 + i + 8 * half;
      const int n = q0 + r;
      const size_t off = ((size_t)(b * N_ + n)) * INNER_ + h * DH_ + cb * 16 + nc;
      ao[off] = f2bf(o[cb][i] / ssum[i]);
    }
}

// ---------------------------------------------------------------------------
// Kernel 4: out = ao(8192x1024 bf16) @ Wout(1024x1024 f32->bf16) -> f32 d_out.
// ---------------------------------------------------------------------------
#define LDA4 40   // 32 + 8 -> 80B rows (16B aligned for vector A loads)
#define LDB4 66

__global__ __launch_bounds__(128) void out_gemm_kernel(
    const __bf16* __restrict__ A, const float* __restrict__ W,
    float* __restrict__ out) {
  __shared__ __bf16 As[64 * LDA4];
  __shared__ __bf16 Bs[32 * LDB4];
  const int tid = threadIdx.x, lane = tid & 31, wave = tid >> 5;
  const int m0 = blockIdx.y * 64, n0 = blockIdx.x * 64;

  v8f acc[4] = {v8f{}, v8f{}, v8f{}, v8f{}};

  for (int k0 = 0; k0 < INNER_; k0 += 32) {
    if (k0 + 32 < INNER_) {
      const int ia = (tid * 8) >> 5, ib = (tid * 4) >> 6;
      __builtin_prefetch(A + (size_t)(m0 + ia) * INNER_ + k0 + 32, 0, 1);
      __builtin_prefetch(W + (size_t)(k0 + 32 + ib) * DIM_ + n0, 0, 1);
    }
    // stage A: 64x32 bf16 (16B vector copies)
#pragma unroll
    for (int i = 0; i < 2; ++i) {
      const int idx = (tid + i * 128) * 8;
      const int r = idx >> 5, c = idx & 31;
      *(uint4*)&As[r * LDA4 + c] =
          *(const uint4*)&A[(size_t)(m0 + r) * INNER_ + k0 + c];
    }
    // stage B: 32x64 f32 -> bf16
#pragma unroll
    for (int i = 0; i < 4; ++i) {
      const int idx = (tid + i * 128) * 4;
      const int r = idx >> 6, c = idx & 63;
      const float4 v = *(const float4*)(W + (size_t)(k0 + r) * DIM_ + n0 + c);
      __bf16* d = &Bs[r * LDB4 + c];
      d[0] = f2bf(v.x); d[1] = f2bf(v.y); d[2] = f2bf(v.z); d[3] = f2bf(v.w);
    }
    __syncthreads();
    const v16bf a = load_a_frag<LDA4>(&As[(wave * 16) * LDA4], lane);
#pragma unroll
    for (int cb = 0; cb < 4; ++cb) {
      const v16bf b = load_b_frag<LDB4>(&Bs[cb * 16], lane);
      acc[cb] = wmma_bf16(a, b, acc[cb]);
    }
    __syncthreads();
  }

  const int half = lane >> 4, nc = lane & 15;
#pragma unroll
  for (int cb = 0; cb < 4; ++cb)
#pragma unroll
    for (int i = 0; i < 8; ++i) {
      const int gr = m0 + wave * 16 + i + half * 8;
      const int gc = n0 + cb * 16 + nc;
      out[(size_t)gr * DIM_ + gc] = acc[cb][i];
    }
}

// ---------------------------------------------------------------------------
extern "C" void kernel_launch(void* const* d_in, const int* in_sizes, int n_in,
                              void* d_out, int out_size, void* d_ws,
                              size_t ws_size, hipStream_t stream) {
  (void)in_sizes; (void)n_in; (void)out_size; (void)ws_size;
  const float* x    = (const float*)d_in[0];
  const float* sinT = (const float*)d_in[1];
  const float* cosT = (const float*)d_in[2];
  const float* Wqkv = (const float*)d_in[3];
  const float* Wout = (const float*)d_in[4];
  float* out = (float*)d_out;

  // workspace layout (bf16): qh | kh | vh | ao, each BH*N*DH = B*N*INNER elems
  const size_t seg = (size_t)BH_ * N_ * DH_;   // 8,388,608 elems (16 MB each)
  char* base = (char*)d_ws;
  __bf16* qh = (__bf16*)(base);
  __bf16* kh = (__bf16*)(base + 2 * seg);
  __bf16* vh = (__bf16*)(base + 4 * seg);
  __bf16* ao = (__bf16*)(base + 6 * seg);

  dim3 g1(3 * INNER_ / 64, (B_ * N_) / 64);
  qkv_gemm_kernel<<<g1, 128, 0, stream>>>(x, Wqkv, qh, kh, vh);

  const int rope_total = 2 * BH_ * N_ * (ROT_ / 2);
  rope_kernel<<<(rope_total + 255) / 256, 256, 0, stream>>>(qh, kh, sinT, cosT);

  dim3 g3(N_ / 64, BH_);
  attn_kernel<<<g3, 128, 0, stream>>>(qh, kh, vh, ao);

  dim3 g4(DIM_ / 64, (B_ * N_) / 64);
  out_gemm_kernel<<<g4, 128, 0, stream>>>(ao, Wout, out);
}
